// EnhancedGNNEncoder_65481071395098
// MI455X (gfx1250) — matrix operand order repaired and checked by reading, not compiled, not run on hardware
//
#include <hip/hip_runtime.h>
#include <hip/hip_bf16.h>

// ---------------------------------------------------------------------------
// EnhancedGNNEncoder (GCN-VAE encoder) for MI455X / gfx1250, wave32 + WMMA.
//
//   deg/dis -> pack(x->f16 N x 96) -> GEMM1(WMMA) -> scatter -> BN+ReLU(f16)
//           -> GEMM2(WMMA) -> scatter -> BN+ReLU(f16)
//           -> GEMM mu / GEMM logstd (WMMA) -> scatter into d_out -> +bias
//
// GEMM fragments are loaded as contiguous b128 f16 runs (A pre-padded,
// B pre-packed into the per-lane WMMA fragment layout): hot loop is
// 4x global_load_b128 + 1x v_wmma per 16x16x32 step, fully unrolled.
// Edge scatter (the bandwidth/issue bottleneck) is float4-vectorized:
// b128 gathers + 4x global_atomic_add_f32, indices amortized over 4 feats.
// b1/b2 cancel under BatchNorm and are dropped.
// ---------------------------------------------------------------------------

typedef __attribute__((ext_vector_type(16))) _Float16 v16h;
typedef __attribute__((ext_vector_type(8)))  _Float16 v8h;
typedef __attribute__((ext_vector_type(8)))  float    v8f;
typedef __attribute__((ext_vector_type(4)))  float    v4f;

#define IN_DIM 81
#define KPAD1  96
#define HID    64
#define LAT    32

// --------------------------- degree / normalization -------------------------

__global__ void deg_kernel(const int* __restrict__ dst, float* __restrict__ deg, int E) {
    int e = blockIdx.x * blockDim.x + threadIdx.x;
    if (e < E) unsafeAtomicAdd(&deg[dst[e]], 1.0f);
}

__global__ void dis_kernel(float* __restrict__ deg, int N) {
    int n = blockIdx.x * blockDim.x + threadIdx.x;
    if (n < N) deg[n] = rsqrtf(deg[n] + 1.0f);   // +1 self-loop; deg >= 1 always
}

// ------------------------------ packing kernels -----------------------------

// x (f32, N x K) -> Ah (f16, N x KPAD) zero-padded. One block per row.
__global__ void pack_a_kernel(const float* __restrict__ X, _Float16* __restrict__ Ah,
                              int K, int KP) {
    int row = blockIdx.x;
    int j = threadIdx.x;
    if (j < KP)
        Ah[(long)row * KP + j] = (j < K) ? (_Float16)X[(long)row * K + j] : (_Float16)0.0f;
}

// W (f32, K x M) -> Bp packed per-lane fragment layout:
//   Bp[((c*tilesM + ct)*32 + lane)*16 + t] = W[c*32 + (lane>>4)*16 + t, ct*16 + (lane&15)]
__global__ void pack_w_kernel(const float* __restrict__ W, _Float16* __restrict__ Bp,
                              int K, int M) {
    int i = blockIdx.x * blockDim.x + threadIdx.x;
    int tilesM = M >> 4;
    int chunks = (K + 31) >> 5;
    if (i >= chunks * tilesM * 512) return;
    int t    = i & 15;
    int lane = (i >> 4) & 31;
    int ct   = (i >> 9) % tilesM;
    int c    = (i >> 9) / tilesM;
    int k   = c * 32 + (lane >> 4) * 16 + t;
    int col = ct * 16 + (lane & 15);
    Bp[i] = (k < K) ? (_Float16)W[(long)k * M + col] : (_Float16)0.0f;
}

// ------------------------------- WMMA GEMM ----------------------------------
// C[N,M] = A[N,KPAD] @ W[K,M] with A f16 row-major (zero-padded) and B
// pre-packed. One wave per 16x16 C tile; grid sized exactly (EXEC all-ones).

template <int KP, int M>
__global__ __launch_bounds__(128)
void wmma_gemm_kernel(const _Float16* __restrict__ A,
                      const _Float16* __restrict__ Bp,
                      float* __restrict__ C, int N) {
    constexpr int TILES_M = M / 16;
    const int wave = threadIdx.x >> 5;
    const int lane = threadIdx.x & 31;
    const int tile = blockIdx.x * 4 + wave;     // 4 waves/block, exact grid
    const int rt = tile / TILES_M;
    const int ct = tile % TILES_M;
    const int row0 = rt << 4;
    const int col0 = ct << 4;
    const int g    = lane >> 4;                 // 0: lanes 0-15, 1: lanes 16-31
    const int lmod = lane & 15;

    const _Float16* ap = A + (long)(row0 + lmod) * KP + g * 16;
    const _Float16* bp = Bp + ((long)ct * 32 + lane) * 16;

    v8f acc = {};
#pragma unroll
    for (int c = 0; c < KP / 32; ++c) {
        v8h a0 = *(const v8h*)(ap + c * 32);        // K = kk + g*16 + 0..7
        v8h a1 = *(const v8h*)(ap + c * 32 + 8);    // K = kk + g*16 + 8..15
        v8h b0 = *(const v8h*)(bp + (long)c * TILES_M * 512);
        v8h b1 = *(const v8h*)(bp + (long)c * TILES_M * 512 + 8);
        v16h a = __builtin_shufflevector(a0, a1, 0,1,2,3,4,5,6,7,8,9,10,11,12,13,14,15);
        v16h b = __builtin_shufflevector(b0, b1, 0,1,2,3,4,5,6,7,8,9,10,11,12,13,14,15);
        acc = __builtin_amdgcn_wmma_f32_16x16x32_f16(
            false, a, false, b, (short)0, acc, false, false);
    }
#pragma unroll
    for (int i = 0; i < 8; ++i) {
        long r = row0 + i + g * 8;                  // C/D layout
        C[r * M + col0 + lmod] = acc[i];
    }
}

// ------------------------- GCN gather/scale/scatter -------------------------
// out[dst] += H[src] * dis[src]*dis[dst]; edges [0,E) real, [E,E+N) self-loops.
// One thread per (edge, 4-feature group): b128 gather + 4x atomic_add_f32;
// H rows are L2-resident (25.6MB << 192MB L2), edge list streams from HBM.

__global__ void gcn_scatter_kernel(const float* __restrict__ H,
                                   const float* __restrict__ dis,
                                   const int* __restrict__ src,
                                   const int* __restrict__ dst,
                                   float* __restrict__ out,
                                   int E, int N, int dshift) {
    const int gshift = dshift - 2;               // groups of 4 features
    long i = (long)blockIdx.x * blockDim.x + threadIdx.x;
    long total = (long)(E + N) << gshift;
    if (i >= total) return;
    int e  = (int)(i >> gshift);
    int d4 = (int)(i & ((1 << gshift) - 1)) << 2;
    int s, t;
    if (e < E) { s = src[e]; t = dst[e]; }
    else       { s = t = e - E; }
    float nrm = dis[s] * dis[t];
    v4f h = *(const v4f*)(H + ((long)s << dshift) + d4);
    float* o = out + ((long)t << dshift) + d4;
    unsafeAtomicAdd(o + 0, h.x * nrm);
    unsafeAtomicAdd(o + 1, h.y * nrm);
    unsafeAtomicAdd(o + 2, h.z * nrm);
    unsafeAtomicAdd(o + 3, h.w * nrm);
}

// ------------------------------- BatchNorm ----------------------------------

__global__ void bn_stats_kernel(const float* __restrict__ H,
                                float* __restrict__ stats, int N, int dshift) {
    int g = blockIdx.x * blockDim.x + threadIdx.x;
    int D = 1 << dshift;
    int c  = g & (D - 1);
    int r0 = g >> dshift;
    int stride = (gridDim.x * blockDim.x) >> dshift;
    float s = 0.0f, s2 = 0.0f;
    for (int r = r0; r < N; r += stride) {
        float v = H[((long)r << dshift) + c];
        s += v; s2 += v * v;
    }
    unsafeAtomicAdd(&stats[c], s);
    unsafeAtomicAdd(&stats[D + c], s2);
}

// Normalize + ReLU, emit f16 (next GEMM's A operand).
__global__ void bn_apply_relu_kernel(const float* __restrict__ Hin,
                                     _Float16* __restrict__ Hout,
                                     const float* __restrict__ stats,
                                     const float* __restrict__ gamma,
                                     const float* __restrict__ beta,
                                     int N, int dshift) {
    long i = (long)blockIdx.x * blockDim.x + threadIdx.x;
    int D = 1 << dshift;
    if (i >= ((long)N << dshift)) return;
    int c = (int)(i & (D - 1));
    float mean = stats[c] / (float)N;
    float var  = stats[D + c] / (float)N - mean * mean;
    float rstd = rsqrtf(var + 1e-5f);
    float y = (Hin[i] - mean) * rstd * gamma[c] + beta[c];
    Hout[i] = (_Float16)(y > 0.0f ? y : 0.0f);
}

__global__ void bias_add_kernel(float* __restrict__ out,
                                const float* __restrict__ b, int N, int lshift) {
    long i = (long)blockIdx.x * blockDim.x + threadIdx.x;
    if (i < ((long)N << lshift)) out[i] += b[i & ((1 << lshift) - 1)];
}

// --------------------------------- launch -----------------------------------

extern "C" void kernel_launch(void* const* d_in, const int* in_sizes, int n_in,
                              void* d_out, int out_size, void* d_ws, size_t ws_size,
                              hipStream_t stream) {
    const float* x     = (const float*)d_in[0];
    const int*   ei    = (const int*)d_in[1];
    const float* W1    = (const float*)d_in[2];
    const float* g1    = (const float*)d_in[4];
    const float* beta1 = (const float*)d_in[5];
    const float* W2    = (const float*)d_in[6];
    const float* g2    = (const float*)d_in[8];
    const float* beta2 = (const float*)d_in[9];
    const float* Wmu   = (const float*)d_in[10];
    const float* bmu   = (const float*)d_in[11];
    const float* Wls   = (const float*)d_in[12];
    const float* bls   = (const float*)d_in[13];

    const int N = in_sizes[0] / IN_DIM;   // 100000 (multiple of 16)
    const int E = in_sizes[1] / 2;        // 3200000
    const int* src = ei;
    const int* dst = ei + E;

    // workspace layout (all regions 16B aligned)
    char* p = (char*)d_ws;
    float* stats = (float*)p;              p += 512;
    float* dis   = (float*)p;              p += (size_t)N * 4;
    float* bufA  = (float*)p;              p += (size_t)N * HID * 4;
    float* bufB  = (float*)p;              p += (size_t)N * HID * 4;
    _Float16* Ah = (_Float16*)p;           p += (size_t)N * KPAD1 * 2;  // reused as Hh
    _Float16* Bp1  = (_Float16*)p;         p += 6144 * 2;  // 3 chunks * 4 tiles * 512
    _Float16* Bp2  = (_Float16*)p;         p += 4096 * 2;  // 2 * 4 * 512
    _Float16* Bpmu = (_Float16*)p;         p += 2048 * 2;  // 2 * 2 * 512
    _Float16* Bpls = (_Float16*)p;         p += 2048 * 2;
    _Float16* Hh = Ah;                     // f16 activations (N*64 <= N*96 slot)
    float* out = (float*)d_out;

    const int T = 256;
    auto blocksFor = [](long n, int t) { return (unsigned)((n + t - 1) / t); };

    const int tiles64 = (N / 16) * (HID / 16);   // 25000 -> 6250 blocks of 4 waves
    const int tiles32 = (N / 16) * (LAT / 16);   // 12500 -> 3125 blocks
    const long sc64 = (long)(E + N) << 4;        // (edge, 4-feat) groups, D=64
    const long sc32 = (long)(E + N) << 3;        // D=32

    // 0) pack all weights into WMMA fragment layout
    pack_w_kernel<<<blocksFor(3 * 4 * 512, T), T, 0, stream>>>(W1,  Bp1,  IN_DIM, HID);
    pack_w_kernel<<<blocksFor(2 * 4 * 512, T), T, 0, stream>>>(W2,  Bp2,  HID,    HID);
    pack_w_kernel<<<blocksFor(2 * 2 * 512, T), T, 0, stream>>>(Wmu, Bpmu, HID,    LAT);
    pack_w_kernel<<<blocksFor(2 * 2 * 512, T), T, 0, stream>>>(Wls, Bpls, HID,    LAT);

    // 1) degree -> dis = rsqrt(deg+1)
    hipMemsetAsync(dis, 0, (size_t)N * sizeof(float), stream);
    deg_kernel<<<blocksFor(E, T), T, 0, stream>>>(dst, dis, E);
    dis_kernel<<<blocksFor(N, T), T, 0, stream>>>(dis, N);

    // 2) layer 1: Ah = pad(x) ; bufA = Ah @ W1 ; conv -> bufB ; BN+ReLU -> Hh
    pack_a_kernel<<<N, 128, 0, stream>>>(x, Ah, IN_DIM, KPAD1);
    wmma_gemm_kernel<KPAD1, HID><<<tiles64 / 4, 128, 0, stream>>>(Ah, Bp1, bufA, N);
    hipMemsetAsync(bufB, 0, (size_t)N * HID * sizeof(float), stream);
    gcn_scatter_kernel<<<blocksFor(sc64, T), T, 0, stream>>>(bufA, dis, src, dst, bufB, E, N, 6);
    hipMemsetAsync(stats, 0, 128 * sizeof(float), stream);
    bn_stats_kernel<<<256, 256, 0, stream>>>(bufB, stats, N, 6);
    bn_apply_relu_kernel<<<blocksFor((long)N << 6, T), T, 0, stream>>>(
        bufB, Hh, stats, g1, beta1, N, 6);

    // 3) layer 2: bufB = Hh @ W2 ; conv -> bufA ; BN+ReLU -> Hh
    wmma_gemm_kernel<HID, HID><<<tiles64 / 4, 128, 0, stream>>>(Hh, Bp2, bufB, N);
    hipMemsetAsync(bufA, 0, (size_t)N * HID * sizeof(float), stream);
    gcn_scatter_kernel<<<blocksFor(sc64, T), T, 0, stream>>>(bufB, dis, src, dst, bufA, E, N, 6);
    hipMemsetAsync(stats, 0, 128 * sizeof(float), stream);
    bn_stats_kernel<<<256, 256, 0, stream>>>(bufA, stats, N, 6);
    bn_apply_relu_kernel<<<blocksFor((long)N << 6, T), T, 0, stream>>>(
        bufA, Hh, stats, g2, beta2, N, 6);

    // 4) heads: Hmu = Hh @ Wmu ; Hls = Hh @ Wls   (into bufA halves, f32)
    float* Hmu = bufA;
    float* Hls = bufA + (size_t)N * LAT;
    wmma_gemm_kernel<HID, LAT><<<tiles32 / 4, 128, 0, stream>>>(Hh, Bpmu, Hmu, N);
    wmma_gemm_kernel<HID, LAT><<<tiles32 / 4, 128, 0, stream>>>(Hh, Bpls, Hls, N);

    // 5) final convs straight into d_out (mu | logstd), then biases
    hipMemsetAsync(out, 0, (size_t)N * 2 * LAT * sizeof(float), stream);
    gcn_scatter_kernel<<<blocksFor(sc32, T), T, 0, stream>>>(Hmu, dis, src, dst, out, E, N, 5);
    gcn_scatter_kernel<<<blocksFor(sc32, T), T, 0, stream>>>(
        Hls, dis, src, dst, out + (size_t)N * LAT, E, N, 5);
    bias_add_kernel<<<blocksFor((long)N << 5, T), T, 0, stream>>>(out, bmu, N, 5);
    bias_add_kernel<<<blocksFor((long)N << 5, T), T, 0, stream>>>(
        out + (size_t)N * LAT, bls, N, 5);
}
